// multihead_self_attention_19086834663473
// MI455X (gfx1250) — matrix operand order repaired and checked by reading, not compiled, not run on hardware
//
#include <hip/hip_runtime.h>

// MI455X / gfx1250, wave32. All matrix math through v_wmma_f32_16x16x32_f16
// (f16 inputs, fp32 accumulate).

typedef __attribute__((ext_vector_type(16))) _Float16 v16h;
typedef __attribute__((ext_vector_type(8)))  float    v8f;

#define DMODEL 1024
#define SEQ    2048
#define NB     2
#define NH     16
#define DK     64
#define MROWS  (NB * SEQ)        // 4096 rows for the projection GEMMs

union Frag16 {
    v16h h;
    unsigned int u[8];
    _Float16 e[16];
};

__device__ __forceinline__ v8f wmma_f16(const Frag16& a, const Frag16& b, v8f c) {
    return __builtin_amdgcn_wmma_f32_16x16x32_f16(
        /*neg_a=*/false, a.h, /*neg_b=*/false, b.h,
        /*c_mod=*/(short)0, c, /*reuse_a=*/false, /*reuse_b=*/false);
}

// ---------------------------------------------------------------- fp32 -> f16
__global__ void cvt_f32_to_f16(const float* __restrict__ in,
                               _Float16* __restrict__ out, int n) {
    int i = blockIdx.x * blockDim.x + threadIdx.x;
    if (i < n) out[i] = (_Float16)in[i];
}

// ---------------------------------------------------------------- WMMA GEMM
// C[m,n] = sum_k A[m,k] * W[n,k]   (A: [M,1024] f16 row-major, W: [1024,1024])
// MODE 0: f16 out, layout [B,H,S,DK]   (Q, K)
// MODE 1: f16 out, layout [B,H,DK,S]   (V transposed for the PV WMMA)
// MODE 2: f32 out, row-major [M,1024]  (final projection -> d_out)
template <int MODE>
__global__ __launch_bounds__(256) void gemm_wmma(
    const _Float16* __restrict__ A, const _Float16* __restrict__ W,
    void* __restrict__ Out) {
    const int lane = threadIdx.x & 31;
    const int wave = threadIdx.x >> 5;
    const int g    = lane >> 4;       // 16-lane half
    const int l15  = lane & 15;
    const int nbase = blockIdx.x * 64;                 // 4 N-tiles per wave
    const int mbase = blockIdx.y * 128 + wave * 16;    // 8 waves stacked in M

    const unsigned int* A32 = (const unsigned int*)A;
    const unsigned int* W32 = (const unsigned int*)W;
    const int ld32 = DMODEL / 2;      // 512 dwords per row
    const int arow = mbase + l15;     // A-fragment row for this lane

    v8f acc[4];
    for (int i = 0; i < 4; ++i) acc[i] = {};

    for (int k0 = 0; k0 < DMODEL; k0 += 32) {
        if (k0 + 32 < DMODEL) {   // prefetch next K slab (global_prefetch_b8)
            __builtin_prefetch(&A32[arow * ld32 + ((k0 + 32) >> 1)], 0, 0);
            __builtin_prefetch(&W32[(nbase + l15) * ld32 + ((k0 + 32) >> 1)], 0, 0);
        }
        // A fragment: 16-bit A-matrix 16x32, contiguous k = 16*g + s (matches B)
        Frag16 af;
#pragma unroll
        for (int j = 0; j < 8; ++j) {
            int kb = 2 * j + 16 * g;
            af.u[j] = A32[arow * ld32 + ((k0 + kb) >> 1)];
        }
#pragma unroll
        for (int nt = 0; nt < 4; ++nt) {
            // B fragment: 32x16 layout; B[k][n] = W[n][k] -> contiguous dwords
            Frag16 bf;
            const int ncol = nbase + nt * 16 + l15;
#pragma unroll
            for (int j = 0; j < 8; ++j)
                bf.u[j] = W32[ncol * ld32 + (k0 >> 1) + j + 8 * g];
            acc[nt] = wmma_f16(af, bf, acc[nt]);
        }
    }

#pragma unroll
    for (int nt = 0; nt < 4; ++nt) {
#pragma unroll
        for (int r = 0; r < 8; ++r) {      // C layout: M = r + 8*g, N = l15
            const int m = mbase + r + 8 * g;
            const int n = nbase + nt * 16 + l15;
            const float v = acc[nt][r];
            if (MODE == 2) {
                ((float*)Out)[(size_t)m * DMODEL + n] = v;
            } else {
                const int b = m >> 11, s = m & (SEQ - 1);
                const int h = n >> 6,  c = n & (DK - 1);
                size_t idx;
                if (MODE == 0) idx = ((size_t)(b * NH + h) * SEQ + s) * DK + c;
                else           idx = ((size_t)(b * NH + h) * DK + c) * SEQ + s;
                ((_Float16*)Out)[idx] = (_Float16)v;
            }
        }
    }
}

// ------------------------------------------------------- causal flash attention
// One wave per 16-row query tile; kv tiles of 32 columns.
// Q,K: [B,H,S,DK] f16.  Vt: [B,H,DK,S] f16.  O: [B,S,D] f16 (heads concatenated).
__global__ __launch_bounds__(128) void attn_fwd(
    const _Float16* __restrict__ Q, const _Float16* __restrict__ K,
    const _Float16* __restrict__ Vt, _Float16* __restrict__ O) {
    __shared__ _Float16 pbuf[4][16 * 32];   // per-wave P staging (C->A transpose)

    const int lane = threadIdx.x & 31;
    const int wave = threadIdx.x >> 5;
    const int g    = lane >> 4;
    const int l15  = lane & 15;

    const int bh    = blockIdx.x >> 5;            // 32 blocks per (b,h)
    const int qt    = (blockIdx.x & 31) * 4 + wave;
    const int qbase = qt * 16;
    const int b = bh >> 4, h = bh & 15;

    const unsigned int* Q32 = (const unsigned int*)(Q + (size_t)bh * SEQ * DK);
    const unsigned int* K32 = (const unsigned int*)(K + (size_t)bh * SEQ * DK);
    const unsigned int* V32 = (const unsigned int*)(Vt + (size_t)bh * DK * SEQ);
    _Float16* pw = pbuf[wave];

    // Q fragments for the two dk-chunks of 32, kept in registers for the loop.
    Frag16 qf[2];
    {
        const int row = qbase + l15;
#pragma unroll
        for (int kc = 0; kc < 2; ++kc)
#pragma unroll
            for (int j = 0; j < 8; ++j) {
                int kb = 2 * j + 16 * g;
                qf[kc].u[j] = Q32[row * (DK / 2) + ((kc * 32 + kb) >> 1)];
            }
    }

    v8f acc[4];
    for (int i = 0; i < 4; ++i) acc[i] = {};
    float mrow[8], lrow[8];
#pragma unroll
    for (int r = 0; r < 8; ++r) { mrow[r] = -1e30f; lrow[r] = 0.0f; }

    const int nIter = ((qbase + 15) >> 5) + 1;
    for (int it = 0; it < nIter; ++it) {
        const int jbase = it * 32;

        // ---- scores S = Q K^T : two 16-wide column tiles, 2 WMMAs each
        v8f sc[2];
#pragma unroll
        for (int ct = 0; ct < 2; ++ct) {
            v8f c = {};
#pragma unroll
            for (int kc = 0; kc < 2; ++kc) {
                Frag16 bf;
                const int tok = jbase + ct * 16 + l15;
#pragma unroll
                for (int j = 0; j < 8; ++j)
                    bf.u[j] = K32[tok * (DK / 2) + kc * 16 + j + 8 * g];
                c = wmma_f16(qf[kc], bf, c);
            }
            sc[ct] = c;
        }

        // ---- scale, causal mask (-10000 like the reference), online softmax
        float alpha[8];
#pragma unroll
        for (int r = 0; r < 8; ++r) {
            const int row = qbase + r + 8 * g;
            float s0 = sc[0][r] * 0.125f;            // 1/sqrt(64)
            float s1 = sc[1][r] * 0.125f;
            if (jbase + l15 > row)      s0 = -10000.0f;
            if (jbase + 16 + l15 > row) s1 = -10000.0f;

            float mx = fmaxf(s0, s1);                // row-max across 16 lanes
            mx = fmaxf(mx, __shfl_xor(mx, 1));
            mx = fmaxf(mx, __shfl_xor(mx, 2));
            mx = fmaxf(mx, __shfl_xor(mx, 4));
            mx = fmaxf(mx, __shfl_xor(mx, 8));
            const float mnew = fmaxf(mrow[r], mx);
            const float a  = __expf(mrow[r] - mnew);
            const float p0 = __expf(s0 - mnew);
            const float p1 = __expf(s1 - mnew);
            float rs = p0 + p1;                      // row-sum of new probs
            rs += __shfl_xor(rs, 1);
            rs += __shfl_xor(rs, 2);
            rs += __shfl_xor(rs, 4);
            rs += __shfl_xor(rs, 8);
            lrow[r] = lrow[r] * a + rs;
            mrow[r] = mnew;
            alpha[r] = a;
            // stage P (f16) to LDS in [qrow][kvcol] order
            pw[(r + 8 * g) * 32 + l15]      = (_Float16)p0;
            pw[(r + 8 * g) * 32 + 16 + l15] = (_Float16)p1;
        }
        asm volatile("s_wait_dscnt 0" ::: "memory");   // LDS writes visible wave-wide

        // ---- P fragment in A layout (16x32)
        Frag16 pf;
#pragma unroll
        for (int hh = 0; hh < 16; ++hh) {
            const int kk = hh + 16 * g;
            pf.e[hh] = pw[l15 * 32 + kk];
        }

        // ---- O += P V : 4 dk-tiles of 16, B fragments from Vt (contiguous)
#pragma unroll
        for (int nt = 0; nt < 4; ++nt) {
#pragma unroll
            for (int r = 0; r < 8; ++r) acc[nt][r] *= alpha[r];
            Frag16 vf;
#pragma unroll
            for (int j = 0; j < 8; ++j)
                vf.u[j] = V32[(nt * 16 + l15) * (SEQ / 2) + (jbase >> 1) + j + 8 * g];
            acc[nt] = wmma_f16(pf, vf, acc[nt]);
        }
    }

    // ---- normalize and store O (f16, heads concatenated along feature dim)
#pragma unroll
    for (int nt = 0; nt < 4; ++nt)
#pragma unroll
        for (int r = 0; r < 8; ++r) {
            const int row = qbase + r + 8 * g;
            const int col = h * DK + nt * 16 + l15;
            const float v = acc[nt][r] / lrow[r];
            O[((size_t)(b * SEQ + row)) * DMODEL + col] = (_Float16)v;
        }
}

// ------------------------------------------------------------------ launcher
extern "C" void kernel_launch(void* const* d_in, const int* in_sizes, int n_in,
                              void* d_out, int out_size, void* d_ws, size_t ws_size,
                              hipStream_t stream) {
    const float* wq = (const float*)d_in[0];
    const float* wk = (const float*)d_in[1];
    const float* wv = (const float*)d_in[2];
    const float* wo = (const float*)d_in[3];
    const float* x  = (const float*)d_in[4];
    float* out = (float*)d_out;

    _Float16* ws = (_Float16*)d_ws;
    size_t off = 0;
    _Float16* xh  = ws + off; off += (size_t)MROWS * DMODEL;     // 4M
    _Float16* wqh = ws + off; off += (size_t)DMODEL * DMODEL;    // 1M
    _Float16* wkh = ws + off; off += (size_t)DMODEL * DMODEL;
    _Float16* wvh = ws + off; off += (size_t)DMODEL * DMODEL;
    _Float16* woh = ws + off; off += (size_t)DMODEL * DMODEL;
    _Float16* Qh  = ws + off; off += (size_t)NB * NH * SEQ * DK; // 4M
    _Float16* Kh  = ws + off; off += (size_t)NB * NH * SEQ * DK;
    _Float16* Vth = ws + off; off += (size_t)NB * NH * SEQ * DK;
    _Float16* Oh  = ws + off; off += (size_t)MROWS * DMODEL;     // 4M

    const int nx = MROWS * DMODEL;
    const int nw = DMODEL * DMODEL;
    cvt_f32_to_f16<<<(nx + 255) / 256, 256, 0, stream>>>(x,  xh,  nx);
    cvt_f32_to_f16<<<(nw + 255) / 256, 256, 0, stream>>>(wq, wqh, nw);
    cvt_f32_to_f16<<<(nw + 255) / 256, 256, 0, stream>>>(wk, wkh, nw);
    cvt_f32_to_f16<<<(nw + 255) / 256, 256, 0, stream>>>(wv, wvh, nw);
    cvt_f32_to_f16<<<(nw + 255) / 256, 256, 0, stream>>>(wo, woh, nw);

    dim3 ggrid(DMODEL / 64, MROWS / 128);  // (16, 32)
    gemm_wmma<0><<<ggrid, 256, 0, stream>>>(xh, wqh, Qh);
    gemm_wmma<0><<<ggrid, 256, 0, stream>>>(xh, wkh, Kh);
    gemm_wmma<1><<<ggrid, 256, 0, stream>>>(xh, wvh, Vth);

    // B*H=32 head-batches, 32 q-tile-groups each (4 waves/block * 16 rows)
    attn_fwd<<<NB * NH * (SEQ / 64), 128, 0, stream>>>(Qh, Kh, Vth, Oh);

    gemm_wmma<2><<<ggrid, 256, 0, stream>>>(Oh, woh, out);
}
